// VQLayer_58884001628201
// MI455X (gfx1250) — matrix operand order repaired and checked by reading, not compiled, not run on hardware
//
#include <hip/hip_runtime.h>
#include <hip/hip_bf16.h>

// ---------------------------------------------------------------------------
// VQ layer for MI455X (gfx1250, wave32, WMMA), fused version.
//   1) transpose x -> bf16 pixel-major (LDS tile transpose)
//   2) convert conv_w / codebook -> bf16, codebook row norms c2[k]
//   3) vq_fused: conv GEMM (v_wmma_f32_16x16x32_bf16) -> encoded f32 out +
//      bf16 e-tile staged in LDS -> distance GEMM (WMMA, A-frags via
//      ds_load_b128) -> argmin (half-wave shfl reduction) -> gather.
//
// argmin_k(||e||^2 + ||c_k||^2 - 2 e.c_k) == argmin_k(c2[k] - 2 e.c_k),
// so per-pixel norms are never computed. out == embeddings in forward.
//
// Workspace (needs ~32.4 MiB):
//   [0,       32 MiB)  x transposed, bf16, row-major [65536][256]
//   [32 MiB, +128 KiB) conv_w bf16 [256][256]
//   [+128K,  +384 KiB) codebook bf16 [512][256]
//   [+384K,  +386 KiB) c2[k] f32 [512]
// ---------------------------------------------------------------------------

typedef __attribute__((ext_vector_type(16))) __bf16        v16bf;
typedef __attribute__((ext_vector_type(8)))  float         v8f;
typedef __attribute__((ext_vector_type(4)))  float         v4f;
typedef __attribute__((ext_vector_type(8)))  unsigned int  v8u;
typedef __attribute__((ext_vector_type(4)))  unsigned int  v4u;

#define NPIX   65536    // B*H*W
#define OUTSZ  16777216 // B*D*H*W

// A-fragment (16x32 bf16, MxK): lanes 0-15 -> M rows 0-15, K at
// kbase+0..15 (v0-7); lanes 16-31 hold kbase+16..31.
// Source row-major [.][256] bf16 -> two 16-byte loads per lane.
__device__ inline v16bf load_a_frag(const __bf16* base, int mbase, int kbase, int lane) {
    int row  = mbase + (lane & 15);
    int koff = kbase + ((lane >> 4) << 4);
    const __bf16* p = base + (size_t)row * 256 + koff;
    v4u lo = *(const v4u*)(p);
    v4u hi = *(const v4u*)(p + 8);
    v8u r = {lo.x, lo.y, lo.z, lo.w, hi.x, hi.y, hi.z, hi.w};
    return __builtin_bit_cast(v16bf, r);
}

// B-fragment (32x16 bf16, KxN): lane L holds column nbase+(L&15);
// lanes 0-15 hold K=kbase..+15, lanes 16-31 hold kbase+16..+31.
// B[k][n] = row_n[k], rows contiguous -> one 32-byte load per lane.
__device__ inline v16bf load_b_frag(const __bf16* rows, int nbase, int kbase, int lane) {
    int col  = nbase + (lane & 15);
    int koff = kbase + ((lane >> 4) << 4);
    return __builtin_bit_cast(v16bf, *(const v8u*)(rows + (size_t)col * 256 + koff));
}

// x (B,C,H,W) f32 -> xt[n][c] bf16 (pixel-major), via LDS tile transpose.
__global__ __launch_bounds__(256) void transpose_x(const float* __restrict__ x,
                                                   __bf16* __restrict__ xt) {
    __shared__ float tile[32][33];
    int b = blockIdx.z, c0 = blockIdx.y * 32, p0 = blockIdx.x * 32;
    int tx = threadIdx.x, ty = threadIdx.y; // 32 x 8
#pragma unroll
    for (int i = 0; i < 4; ++i) {
        int c = c0 + ty + i * 8;
        tile[ty + i * 8][tx] = x[((size_t)(b * 256 + c) << 12) + p0 + tx];
    }
    __syncthreads();
#pragma unroll
    for (int i = 0; i < 4; ++i) {
        int p = p0 + ty + i * 8;
        xt[((size_t)(b * 4096 + p) << 8) + c0 + tx] = (__bf16)tile[tx][ty + i * 8];
    }
}

__global__ __launch_bounds__(256) void prep_w(const float* __restrict__ w,
                                              __bf16* __restrict__ wb) {
    int i = blockIdx.x * 256 + threadIdx.x;
    wb[i] = (__bf16)w[i];
}

// codebook row -> bf16 and c2[k] = ||codebook_k||^2.
__global__ __launch_bounds__(256) void prep_cb(const float* __restrict__ cb,
                                               __bf16* __restrict__ cbb,
                                               float* __restrict__ c2) {
    int k = blockIdx.x, t = threadIdx.x;
    float v = cb[(size_t)k * 256 + t];
    cbb[(size_t)k * 256 + t] = (__bf16)v;
    float s = v * v;
#pragma unroll
    for (int off = 16; off > 0; off >>= 1) s += __shfl_xor(s, off, 32);
    __shared__ float wsum[8];
    if ((t & 31) == 0) wsum[t >> 5] = s;
    __syncthreads();
    if (t == 0) {
        float tot = 0.f;
#pragma unroll
        for (int i = 0; i < 8; ++i) tot += wsum[i];
        c2[k] = tot;
    }
}

// Fused conv GEMM + VQ distance GEMM + argmin + gather. 128 pixels / block.
__global__ __launch_bounds__(256) void vq_fused(const __bf16* __restrict__ xt,
                                                const __bf16* __restrict__ wb,
                                                const float* __restrict__ bias,
                                                const __bf16* __restrict__ cbb,
                                                const float* __restrict__ c2,
                                                const float* __restrict__ cb,
                                                float* __restrict__ enc,
                                                float* __restrict__ out,
                                                float* __restrict__ emb,
                                                int* __restrict__ idxout) {
    // Per-wave 16x256 bf16 e-tile; 8 waves -> 64 KB (WGP has 320 KB).
    __shared__ __align__(16) __bf16 etile[8 * 16 * 256];

    int tid  = threadIdx.x;
    int lane = tid & 31, wave = tid >> 5;
    int mbase = blockIdx.x * 128 + wave * 16;
    int bimg  = mbase >> 12;            // image in batch (tile never crosses b)
    int dcol  = lane & 15;
    int half8 = (lane >> 4) << 3;       // C layout: lanes>=16 hold M rows 8..15
    __bf16* etw = etile + wave * (16 * 256);

    // ---- Phase 1: encoded = xt * W^T + bias -------------------------------
    {
        v16bf a[8];
#pragma unroll
        for (int kk = 0; kk < 8; ++kk) a[kk] = load_a_frag(xt, mbase, kk * 32, lane);

        int p0 = (mbase & 4095) + half8; // first pixel this lane's C rows cover

        for (int nt = 0; nt < 16; ++nt) {
            int nbase = nt * 16;
            v16bf bf[8];
#pragma unroll
            for (int kk = 0; kk < 8; ++kk) bf[kk] = load_b_frag(wb, nbase, kk * 32, lane);
            v8f acc = {};
#pragma unroll
            for (int kk = 0; kk < 8; ++kk)
                acc = __builtin_amdgcn_wmma_f32_16x16x32_bf16(
                    false, a[kk], false, bf[kk], (short)0, acc, false, false);

            int d = nbase + dcol;
            float bv = bias[d];
            v4f s0 = {acc[0] + bv, acc[1] + bv, acc[2] + bv, acc[3] + bv};
            v4f s1 = {acc[4] + bv, acc[5] + bv, acc[6] + bv, acc[7] + bv};
            // 8 consecutive pixels at fixed d -> two b128 stores (B,D,H,W).
            float* ep = enc + ((size_t)(bimg * 256 + d) << 12) + p0;
            *(v4f*)(ep)     = s0;
            *(v4f*)(ep + 4) = s1;
            // bf16 e-tile in LDS, row-major [m_local][256].
#pragma unroll
            for (int r = 0; r < 4; ++r) {
                etw[(half8 + r) * 256 + d]     = (__bf16)s0[r];
                etw[(half8 + 4 + r) * 256 + d] = (__bf16)s1[r];
            }
        }
    }

    // ---- Phase 2: scores = c2[k] - 2 e.c_k, argmin over k -----------------
    v16bf a2[8];
#pragma unroll
    for (int kk = 0; kk < 8; ++kk) a2[kk] = load_a_frag(etw, 0, kk * 32, lane);
    __syncthreads(); // all e-tile reads done -> safe to alias sidx below

    float best[8];
    int   bidx[8];
#pragma unroll
    for (int r = 0; r < 8; ++r) { best[r] = 3.4e38f; bidx[r] = 0; }

    for (int nt = 0; nt < 32; ++nt) {   // K=512 codebook entries, 16 per tile
        int nbase = nt * 16;
        v16bf bf[8];
#pragma unroll
        for (int kk = 0; kk < 8; ++kk) bf[kk] = load_b_frag(cbb, nbase, kk * 32, lane);
        v8f acc = {};
#pragma unroll
        for (int kk = 0; kk < 8; ++kk)
            acc = __builtin_amdgcn_wmma_f32_16x16x32_bf16(
                false, a2[kk], false, bf[kk], (short)0, acc, false, false);

        int n = nbase + dcol;
        float c2n = c2[n];
#pragma unroll
        for (int r = 0; r < 8; ++r) {
            float s = fmaf(-2.0f, acc[r], c2n);
            if (s < best[r]) { best[r] = s; bidx[r] = n; } // strict < -> lowest n
        }
    }

    // argmin across the 16 lanes of each half-wave, index tie-break.
#pragma unroll
    for (int off = 1; off < 16; off <<= 1) {
#pragma unroll
        for (int r = 0; r < 8; ++r) {
            float ob = __shfl_xor(best[r], off, 32);
            int   oi = __shfl_xor(bidx[r], off, 32);
            if (ob < best[r] || (ob == best[r] && oi < bidx[r])) {
                best[r] = ob; bidx[r] = oi;
            }
        }
    }

    int* sidx = (int*)etile; // e-tile is dead; reuse LDS for 128 indices
    if ((lane & 15) == 0) {
#pragma unroll
        for (int r = 0; r < 8; ++r) {
            int ml = wave * 16 + half8 + r;
            idxout[blockIdx.x * 128 + ml] = bidx[r];
            sidx[ml] = bidx[r];
        }
    }
    __syncthreads();

    // ---- Phase 3: gather codebook rows -> out & embeddings ----------------
    // 128 pixels x 256 dims, 2 threads per pixel (contiguous 128-d halves).
    int pl = tid & 127;
    int m  = blockIdx.x * 128 + pl;
    int ci = sidx[pl];
    const float* crow = cb + (size_t)ci * 256;
    __builtin_prefetch(crow, 0, 1);
    int p = m & 4095;
    int dbase = (tid >> 7) << 7; // 0 or 128
#pragma unroll 4
    for (int j = 0; j < 128; j += 4) {
        v4f v = *(const v4f*)(crow + dbase + j);
#pragma unroll
        for (int q = 0; q < 4; ++q) {
            int d = dbase + j + q;
            size_t o = ((size_t)(bimg * 256 + d) << 12) + p;
            out[o] = v[q];   // out == embeddings in forward pass
            emb[o] = v[q];
        }
    }
}

extern "C" void kernel_launch(void* const* d_in, const int* in_sizes, int n_in,
                              void* d_out, int out_size, void* d_ws, size_t ws_size,
                              hipStream_t stream) {
    const float* x        = (const float*)d_in[0];
    const float* conv_w   = (const float*)d_in[1];
    const float* conv_b   = (const float*)d_in[2];
    const float* codebook = (const float*)d_in[3];

    float* out = (float*)d_out;                   // (B,D,H,W)
    float* emb = out + OUTSZ;                     // (B,D,H,W)
    float* enc = out + 2 * (size_t)OUTSZ;         // (B,D,H,W)
    int*   idx = (int*)(out + 3 * (size_t)OUTSZ); // (B,H,W) int32

    char* ws = (char*)d_ws;
    __bf16* xt  = (__bf16*)ws;                                       // 32 MiB
    __bf16* wbb = (__bf16*)(ws + ((size_t)32 << 20));                // 128 KiB
    __bf16* cbb = (__bf16*)(ws + ((size_t)32 << 20) + (128u << 10)); // 256 KiB
    float*  c2  = (float*)(ws + ((size_t)32 << 20) + (384u << 10));  // 2 KiB

    transpose_x<<<dim3(128, 8, 16), dim3(32, 8), 0, stream>>>(x, xt);
    prep_w<<<256, 256, 0, stream>>>(conv_w, wbb);
    prep_cb<<<512, 256, 0, stream>>>(codebook, cbb, c2);
    vq_fused<<<NPIX / 128, 256, 0, stream>>>(xt, wbb, conv_b, cbb, c2, codebook,
                                             enc, out, emb, idx);
}
